// ParallelAttentionLayer_6107443495567
// MI455X (gfx1250) — compile-verified, matrix operand order checked by reading, not run on hardware
//
#include <hip/hip_runtime.h>
#include <hip/hip_bf16.h>
#include <math.h>

// ---------------------------------------------------------------------------
// MI455X (gfx1250) fused parallel-attention layer, compile-only tuning target.
// All GEMMs use v_wmma_f32_16x16x32_f16 (f16 in, f32 accumulate).
// fp32 -> f16 conversion is hoisted out of the GEMM inner loops, and the
// attention kernel streams K tiles through LDS with the Tensor Data Mover
// (tensor_load_to_lds + s_wait_tensorcnt) when the toolchain exposes it.
// ---------------------------------------------------------------------------

#define LL   1024   // queries
#define SS   1024   // keys
#define MM   256    // memory slots
#define NN   8      // batch
#define EE   512    // embed
#define HH   8      // heads
#define HD   64     // head dim
#define NHB  64     // N*H head-batches

typedef __attribute__((ext_vector_type(16))) _Float16 v16h;
typedef __attribute__((ext_vector_type(8)))  _Float16 v8h;
typedef __attribute__((ext_vector_type(4)))  _Float16 v4h;
typedef __attribute__((ext_vector_type(8)))  float    v8f;
typedef __attribute__((ext_vector_type(4)))  float    v4f;
typedef __attribute__((ext_vector_type(4)))  unsigned int u32x4;
typedef __attribute__((ext_vector_type(4)))  int      i32x4;
typedef __attribute__((ext_vector_type(8)))  int      i32x8;

#if __has_builtin(__builtin_amdgcn_tensor_load_to_lds) && \
    __has_builtin(__builtin_amdgcn_s_wait_tensorcnt)
#define CDNA5_TDM 1
#else
#define CDNA5_TDM 0
#endif

#define DEV __device__ __forceinline__

DEV v8f vzero8() {
  v8f c;
#pragma unroll
  for (int i = 0; i < 8; ++i) c[i] = 0.0f;
  return c;
}

DEV v8f wmma_f16(v16h a, v16h b, v8f c) {
  // D = A(16x32 f16) * B(32x16 f16) + C(16x16 f32)
  return __builtin_amdgcn_wmma_f32_16x16x32_f16(
      false, a, false, b, (short)0, c, false, false);
}

// Load a 16x32 f16 A/B fragment from a row-major tile.
// ISA layout: lane = (row%16) + 16*khalf; v16h holds K = khalf*8+{0..7} and
// 16+khalf*8+{0..7}. Works for both A (rows = M) and B (rows = N == B^T rows).
DEV v16h load_frag_h(const _Float16* tile, int stride, int k0, int lane) {
  const int r  = lane & 15;
  const int hf = lane >> 4;
  const _Float16* p = tile + (size_t)r * stride + k0 + hf * 8;
  v8h lo = *(const v8h*)(p);
  v8h hi = *(const v8h*)(p + 16);
  v16h f;
#pragma unroll
  for (int i = 0; i < 8; ++i) { f[i] = lo[i]; f[i + 8] = hi[i]; }
  return f;
}

#if CDNA5_TDM
// Tensor Data Mover: async copy of a 2D f16 tile (rows x elems, row stride in
// elements) from global memory into LDS.  Descriptor packed per CDNA5 ISA
// 8.3/8.4 (group0: count/lds_addr/global_addr/type=2; group1: data_size=2B,
// tensor dims, tile dims, dim0 stride).  Tracked with TENSORcnt.
DEV void tdm_load_tile_f16(const void* gptr, unsigned lds_off,
                           unsigned elems, unsigned rows, unsigned stride) {
  const unsigned long long ga = (unsigned long long)(uintptr_t)gptr;
  u32x4 g0;
  g0[0] = 1u;                                   // count=1 (valid descriptor)
  g0[1] = lds_off;                              // lds_addr (bytes)
  g0[2] = (unsigned)(ga & 0xffffffffu);         // global_addr[31:0]
  g0[3] = (unsigned)((ga >> 32) & 0x01ffffffu)  // global_addr[56:32]
          | (2u << 30);                         // type = 2 ("image")
  i32x8 g1;
  g1[0] = 1 << 16;                              // data_size = 1 -> 2 bytes
  g1[1] = (int)(elems << 16);                   // tensor_dim0[15:0]
  g1[2] = (int)(rows << 16);                    // tensor_dim1[15:0]
  g1[3] = (int)(elems << 16);                   // tile_dim0
  g1[4] = (int)rows;                            // tile_dim1 (tile_dim2 = 0)
  g1[5] = (int)stride;                          // tensor_dim0_stride[31:0]
  g1[6] = 0;
  g1[7] = 0;
  i32x4 gz;
#pragma unroll
  for (int i = 0; i < 4; ++i) gz[i] = 0;
#if __clang_major__ >= 23
  i32x8 gz8;
#pragma unroll
  for (int i = 0; i < 8; ++i) gz8[i] = 0;
  __builtin_amdgcn_tensor_load_to_lds(g0, g1, gz, gz, gz8, 0);
#else
  __builtin_amdgcn_tensor_load_to_lds(g0, g1, gz, gz, 0);
#endif
}
#endif  // CDNA5_TDM

// ---------------------------------------------------------------------------
// Kernel 0: bulk fp32 -> f16 conversion (activations + weights), 4 elem/thread.
// ---------------------------------------------------------------------------
__global__ void k_cvt16(const float* __restrict__ src,
                        _Float16* __restrict__ dst) {
  const int i = (blockIdx.x * 256 + threadIdx.x) * 4;
  v4f s = *(const v4f*)(src + i);
  v4h d;
#pragma unroll
  for (int j = 0; j < 4; ++j) d[j] = (_Float16)s[j];
  *(v4h*)(dst + i) = d;
}

// ---------------------------------------------------------------------------
// Kernel 1: fused QKV projection (+bias, +scale for Q, +rotary for Q/K).
// grid = (EE/16, LL/16, 3*NN); one wave per 16x16 output tile; pure-f16 GEMM.
// Outputs (f16): qh/kh row-major per head-batch [nh][t][d]; v transposed
// [nh][d][s] so it is the B operand of the w*V WMMA with unit-stride K.
// ---------------------------------------------------------------------------
__global__ void __launch_bounds__(32)
k_qkv_rope(const _Float16* __restrict__ xq, const _Float16* __restrict__ xk,
           const _Float16* __restrict__ xv, const _Float16* __restrict__ W16,
           const float* __restrict__ bias, const float* __restrict__ rot_q,
           const float* __restrict__ rot_k, _Float16* __restrict__ qh,
           _Float16* __restrict__ kh, _Float16* __restrict__ vT) {
  const int lane = threadIdx.x;
  const int r = lane & 15, hf = lane >> 4;
  const int e0 = blockIdx.x * 16;
  const int t0 = blockIdx.y * 16;
  const int which = blockIdx.z / NN;  // 0=q, 1=k, 2=v
  const int n = blockIdx.z % NN;

  const _Float16* X = (which == 0) ? xq : (which == 1 ? xk : xv);
  const _Float16* Xtile = X + ((size_t)t0 * NN + n) * EE;       // rows: t
  const _Float16* Wtile = W16 + (size_t)(which * EE + e0) * EE; // rows: e_out

  v8f c = vzero8();
#pragma unroll 4
  for (int k0 = 0; k0 < EE; k0 += 32) {
    v16h a = load_frag_h(Xtile, NN * EE, k0, lane);
    v16h b = load_frag_h(Wtile, EE, k0, lane);
    c = wmma_f16(a, b, c);
  }

  const int col = e0 + r;
  const float bcol = bias[which * EE + col];
  const float scale = (which == 0) ? 0.125f : 1.0f;  // hd^-0.5 = 1/8
  const float* rot = (which == 0) ? rot_q : rot_k;
  const int h = col >> 6, d = col & 63;

#pragma unroll
  for (int i = 0; i < 8; ++i) {
    const int t = t0 + i + hf * 8;
    float v = (c[i] + bcol) * scale;
    if (which < 2) {
      // interleaved rotate-half: pair partner sits in lane^1 of the C layout
      float prt = __shfl_xor(v, 1, 32);
      float x2 = (col & 1) ? prt : -prt;
      const float* rp = rot + (((size_t)n * LL + t) * EE + col) * 2;
      v = v * rp[0] + x2 * rp[1];
    }
    const int nh = n * HH + h;
    if (which == 0)
      qh[((size_t)nh * LL + t) * HD + d] = (_Float16)v;
    else if (which == 1)
      kh[((size_t)nh * SS + t) * HD + d] = (_Float16)v;
    else
      vT[((size_t)nh * HD + d) * SS + t] = (_Float16)v;
  }
}

// ---------------------------------------------------------------------------
// Kernel 2: head-split + f16 convert of memory K/V. k_mem/v_mem are [M,N,E].
// ---------------------------------------------------------------------------
__global__ void k_memcvt(const float* __restrict__ km_in,
                         const float* __restrict__ vm_in,
                         _Float16* __restrict__ kmh,
                         _Float16* __restrict__ vmT) {
  const int idx = blockIdx.x * 256 + threadIdx.x;  // < M*N*E = 1048576
  const int m = idx >> 12;          // / (N*E)
  const int rem = idx & 4095;
  const int n = rem >> 9;           // / E
  const int e = rem & 511;
  const int h = e >> 6, d = e & 63;
  const int nh = n * HH + h;
  kmh[((size_t)nh * MM + m) * HD + d] = (_Float16)km_in[idx];
  vmT[((size_t)nh * HD + d) * MM + m] = (_Float16)vm_in[idx];
}

// ---------------------------------------------------------------------------
// Kernel 3: per-key-column softmax stats over the QUERY axis (reference
// softmaxes axis=2 == L).  Online max/sum over L with WMMA score tiles.
// grid = (SS/16, NHB), one wave per 16-column strip.
// ---------------------------------------------------------------------------
__global__ void __launch_bounds__(32)
k_colstats(const _Float16* __restrict__ qh, const _Float16* __restrict__ kh,
           float* __restrict__ cmax, float* __restrict__ csum) {
  const int lane = threadIdx.x;
  const int nh = blockIdx.y;
  const int s0 = blockIdx.x * 16;

  const _Float16* ktile = kh + ((size_t)nh * SS + s0) * HD;
  const v16h b0 = load_frag_h(ktile, HD, 0, lane);
  const v16h b1 = load_frag_h(ktile, HD, 32, lane);

  float m_run = -3.0e38f, s_run = 0.0f;
  for (int lt = 0; lt < LL; lt += 16) {
    const _Float16* qtile = qh + ((size_t)nh * LL + lt) * HD;
    v16h a0 = load_frag_h(qtile, HD, 0, lane);
    v16h a1 = load_frag_h(qtile, HD, 32, lane);
    v8f c = vzero8();
    c = wmma_f16(a0, b0, c);
    c = wmma_f16(a1, b1, c);
    // per-lane partial over 8 rows, then combine khalf partner (rows +8)
    float tm = c[0];
#pragma unroll
    for (int i = 1; i < 8; ++i) tm = fmaxf(tm, c[i]);
    tm = fmaxf(tm, __shfl_xor(tm, 16, 32));
    const float nm = fmaxf(m_run, tm);
    float p = 0.0f;
#pragma unroll
    for (int i = 0; i < 8; ++i) p += __expf(c[i] - nm);
    p += __shfl_xor(p, 16, 32);
    s_run = s_run * __expf(m_run - nm) + p;
    m_run = nm;
  }
  if (lane < 16) {
    cmax[(size_t)nh * SS + s0 + lane] = m_run;
    csum[(size_t)nh * SS + s0 + lane] = s_run;
  }
}

// ---------------------------------------------------------------------------
// Kernel 4: fused attention. One wave per (nh, 16 query rows).
//  - main branch: K tiles stream through an LDS double buffer via the Tensor
//    Data Mover (async, TENSORcnt) when available; scores -> weights with the
//    precomputed column stats (+1e-8, row sums tracked for S-renorm);
//    w tile round-trips LDS (C->A layout) and WMMAs against V^T.
//  - memory branch: sm tiles -> LDS, masked row softmax over M, wm*Vm WMMA.
//  - sigmoid head-gated combine, f16 output in [l*N+n][e] layout.
// ---------------------------------------------------------------------------
__global__ void __launch_bounds__(32)
k_attn(const _Float16* __restrict__ qh, const _Float16* __restrict__ kh,
       const _Float16* __restrict__ vT, const _Float16* __restrict__ kmh,
       const _Float16* __restrict__ vmT, const float* __restrict__ cmax,
       const float* __restrict__ csum, const float* __restrict__ gate_attn,
       const unsigned char* __restrict__ mem_mask,
       _Float16* __restrict__ attnC) {
  __shared__ _Float16 wbuf[16][32];    // score-weight tile (C->A transpose)
  __shared__ float    smbuf[16][256];  // memory scores
  __shared__ _Float16 wmbuf[16][256];  // memory softmax weights
#if CDNA5_TDM
  __shared__ __align__(128) _Float16 kbuf[2][32 * HD];  // K double buffer
#endif

  const int lane = threadIdx.x;
  const int r = lane & 15, hf = lane >> 4;
  const int nh = blockIdx.y;
  const int l0 = blockIdx.x * 16;
  const int n = nh >> 3, h = nh & 7;

  const _Float16* qtile = qh + ((size_t)nh * LL + l0) * HD;
  const v16h a0 = load_frag_h(qtile, HD, 0, lane);
  const v16h a1 = load_frag_h(qtile, HD, 32, lane);

  v8f accD[4];
  float rowsum[8];
#pragma unroll
  for (int i = 0; i < 4; ++i) accD[i] = vzero8();
#pragma unroll
  for (int i = 0; i < 8; ++i) rowsum[i] = 0.0f;

  // ---- main branch over S in double-tiles of 32 keys ----
#if CDNA5_TDM
  tdm_load_tile_f16(kh + (size_t)nh * SS * HD,
                    (unsigned)(uintptr_t)&kbuf[0][0], HD, 32, HD);
#endif
  for (int sb = 0; sb < SS; sb += 32) {
#if CDNA5_TDM
    const int buf = (sb >> 5) & 1;
    if (sb + 32 < SS) {
      tdm_load_tile_f16(kh + ((size_t)nh * SS + sb + 32) * HD,
                        (unsigned)(uintptr_t)&kbuf[buf ^ 1][0], HD, 32, HD);
      __builtin_amdgcn_s_wait_tensorcnt(1);  // current buffer ready
    } else {
      __builtin_amdgcn_s_wait_tensorcnt(0);
    }
    __syncthreads();
    const _Float16* kt_base = &kbuf[buf][0];
#else
    if (sb + 32 < SS)
      __builtin_prefetch(kh + ((size_t)nh * SS + sb + 32) * HD, 0, 1);
    const _Float16* kt_base = kh + ((size_t)nh * SS + sb) * HD;
#endif
#pragma unroll
    for (int sub = 0; sub < 2; ++sub) {
      const int s0 = sb + sub * 16;
      v16h b0 = load_frag_h(kt_base + sub * 16 * HD, HD, 0, lane);
      v16h b1 = load_frag_h(kt_base + sub * 16 * HD, HD, 32, lane);
      v8f c = vzero8();
      c = wmma_f16(a0, b0, c);
      c = wmma_f16(a1, b1, c);
      const int scol = s0 + r;
      const float cm = cmax[(size_t)nh * SS + scol];
      const float inv = 1.0f / csum[(size_t)nh * SS + scol];
#pragma unroll
      for (int i = 0; i < 8; ++i) {
        float w = __expf(c[i] - cm) * inv + 1e-8f;
        rowsum[i] += w;
        wbuf[i + hf * 8][sub * 16 + r] = (_Float16)w;
      }
    }
    __syncthreads();
    v16h wf = load_frag_h(&wbuf[0][0], 32, 0, lane);
#pragma unroll
    for (int dt = 0; dt < 4; ++dt) {
      const _Float16* vtile = vT + ((size_t)nh * HD + dt * 16) * SS;
      v16h bv = load_frag_h(vtile, SS, sb, lane);
      accD[dt] = wmma_f16(wf, bv, accD[dt]);
    }
    __syncthreads();
  }
  // row sums: reduce across the 16 lanes sharing each row group
#pragma unroll
  for (int i = 0; i < 8; ++i) {
    float s = rowsum[i];
    s += __shfl_xor(s, 1, 32);
    s += __shfl_xor(s, 2, 32);
    s += __shfl_xor(s, 4, 32);
    s += __shfl_xor(s, 8, 32);
    rowsum[i] = s;
  }

  // ---- memory branch: sm = Q*Km^T, mask, row softmax over M=256 ----
  for (int mt = 0; mt < MM; mt += 16) {
    const _Float16* kmt = kmh + ((size_t)nh * MM + mt) * HD;
    v16h b0 = load_frag_h(kmt, HD, 0, lane);
    v16h b1 = load_frag_h(kmt, HD, 32, lane);
    v8f c = vzero8();
    c = wmma_f16(a0, b0, c);
    c = wmma_f16(a1, b1, c);
    const int mcol = mt + r;
    const bool msk = mem_mask[n * MM + mcol] != 0;
#pragma unroll
    for (int i = 0; i < 8; ++i)
      smbuf[i + hf * 8][mcol] = msk ? -1e9f : c[i];
  }
  __syncthreads();
  if (lane < 16) {  // one lane per query row
    float rm = -3.0e38f;
    for (int m = 0; m < MM; ++m) rm = fmaxf(rm, smbuf[lane][m]);
    float rs = 0.0f;
    for (int m = 0; m < MM; ++m) rs += __expf(smbuf[lane][m] - rm);
    const float invr = 1.0f / rs;
    for (int m = 0; m < MM; ++m)
      wmbuf[lane][m] = (_Float16)(__expf(smbuf[lane][m] - rm) * invr);
  }
  __syncthreads();

  v8f accM[4];
#pragma unroll
  for (int i = 0; i < 4; ++i) accM[i] = vzero8();
  for (int kt = 0; kt < MM; kt += 32) {
    v16h wf = load_frag_h(&wmbuf[0][0], 256, kt, lane);
#pragma unroll
    for (int dt = 0; dt < 4; ++dt) {
      const _Float16* vtile = vmT + ((size_t)nh * HD + dt * 16) * MM;
      v16h bv = load_frag_h(vtile, MM, kt, lane);
      accM[dt] = wmma_f16(wf, bv, accM[dt]);
    }
  }

  // ---- gated combine + store in [l*N+n][e] f16 layout ----
  const float g = 1.0f / (1.0f + __expf(-gate_attn[h]));
#pragma unroll
  for (int dt = 0; dt < 4; ++dt) {
#pragma unroll
    for (int i = 0; i < 8; ++i) {
      const int l = l0 + i + hf * 8;
      const int d = dt * 16 + r;
      const float val = g * accM[dt][i] + (1.0f - g) * (accD[dt][i] / rowsum[i]);
      attnC[((size_t)l * NN + n) * EE + h * HD + d] = (_Float16)val;
    }
  }
}

// ---------------------------------------------------------------------------
// Kernel 5: output projection. rows = L*N flattened, cols = E, pure-f16 GEMM.
// ---------------------------------------------------------------------------
__global__ void __launch_bounds__(32)
k_outproj(const _Float16* __restrict__ attnC, const _Float16* __restrict__ Wo16,
          const float* __restrict__ bo, float* __restrict__ out) {
  const int lane = threadIdx.x;
  const int r = lane & 15, hf = lane >> 4;
  const int e0 = blockIdx.x * 16;
  const int r0 = blockIdx.y * 16;
  const _Float16* atile = attnC + (size_t)r0 * EE;
  const _Float16* wtile = Wo16 + (size_t)e0 * EE;
  v8f c = vzero8();
#pragma unroll 4
  for (int k0 = 0; k0 < EE; k0 += 32) {
    v16h a = load_frag_h(atile, EE, k0, lane);
    v16h b = load_frag_h(wtile, EE, k0, lane);
    c = wmma_f16(a, b, c);
  }
  const int col = e0 + r;
  const float bb = bo[col];
#pragma unroll
  for (int i = 0; i < 8; ++i)
    out[(size_t)(r0 + i + hf * 8) * EE + col] = c[i] + bb;
}

// ---------------------------------------------------------------------------
extern "C" void kernel_launch(void* const* d_in, const int* in_sizes, int n_in,
                              void* d_out, int out_size, void* d_ws,
                              size_t ws_size, hipStream_t stream) {
  const float* query  = (const float*)d_in[0];
  const float* key    = (const float*)d_in[1];
  const float* value  = (const float*)d_in[2];
  const float* W_in   = (const float*)d_in[3];
  const float* b_in   = (const float*)d_in[4];
  const float* W_out  = (const float*)d_in[5];
  const float* b_out  = (const float*)d_in[6];
  const float* rot_q  = (const float*)d_in[7];
  const float* rot_k  = (const float*)d_in[8];
  const float* k_mem  = (const float*)d_in[9];
  const float* v_mem  = (const float*)d_in[10];
  const float* gate   = (const float*)d_in[11];
  const unsigned char* mem_mask = (const unsigned char*)d_in[12];
  float* out = (float*)d_out;

  // workspace carve-out (~64 MB)
  char* ws = (char*)d_ws;
  size_t off = 0;
  auto carve = [&](size_t bytes) -> char* {
    char* p = ws + off;
    off += (bytes + 255) & ~(size_t)255;
    return p;
  };
  _Float16* xq16  = (_Float16*)carve((size_t)LL * NN * EE * 2);
  _Float16* xk16  = (_Float16*)carve((size_t)SS * NN * EE * 2);
  _Float16* xv16  = (_Float16*)carve((size_t)SS * NN * EE * 2);
  _Float16* W16   = (_Float16*)carve((size_t)3 * EE * EE * 2);
  _Float16* Wo16  = (_Float16*)carve((size_t)EE * EE * 2);
  _Float16* qh    = (_Float16*)carve((size_t)NHB * LL * HD * 2);
  _Float16* kh    = (_Float16*)carve((size_t)NHB * SS * HD * 2);
  _Float16* vT    = (_Float16*)carve((size_t)NHB * HD * SS * 2);
  _Float16* kmh   = (_Float16*)carve((size_t)NHB * MM * HD * 2);
  _Float16* vmT   = (_Float16*)carve((size_t)NHB * HD * MM * 2);
  float*    cmax  = (float*)carve((size_t)NHB * SS * 4);
  float*    csum  = (float*)carve((size_t)NHB * SS * 4);
  _Float16* attnC = (_Float16*)carve((size_t)LL * NN * EE * 2);

  // one-shot f16 conversions (4 elems/thread, 1024 elems/block)
  k_cvt16<<<dim3((LL * NN * EE) / 1024), 256, 0, stream>>>(query, xq16);
  k_cvt16<<<dim3((SS * NN * EE) / 1024), 256, 0, stream>>>(key, xk16);
  k_cvt16<<<dim3((SS * NN * EE) / 1024), 256, 0, stream>>>(value, xv16);
  k_cvt16<<<dim3((3 * EE * EE) / 1024), 256, 0, stream>>>(W_in, W16);
  k_cvt16<<<dim3((EE * EE) / 1024), 256, 0, stream>>>(W_out, Wo16);

  k_qkv_rope<<<dim3(EE / 16, LL / 16, 3 * NN), 32, 0, stream>>>(
      xq16, xk16, xv16, W16, b_in, rot_q, rot_k, qh, kh, vT);
  k_memcvt<<<dim3((MM * NN * EE) / 256), 256, 0, stream>>>(k_mem, v_mem, kmh,
                                                           vmT);
  k_colstats<<<dim3(SS / 16, NHB), 32, 0, stream>>>(qh, kh, cmax, csum);
  k_attn<<<dim3(LL / 16, NHB), 32, 0, stream>>>(qh, kh, vT, kmh, vmT, cmax,
                                                csum, gate, mem_mask, attnC);
  k_outproj<<<dim3(EE / 16, (LL * NN) / 16), 32, 0, stream>>>(attnC, Wo16,
                                                              b_out, out);
}
